// SNN_1400159338902
// MI455X (gfx1250) — matrix operand order repaired and checked by reading, not compiled, not run on hardware
//
#include <hip/hip_runtime.h>
#include <hip/hip_bf16.h>
#include <math.h>

// ---------------------------------------------------------------------------
// Types for CDNA5 WMMA
// ---------------------------------------------------------------------------
typedef __attribute__((ext_vector_type(16))) __bf16 v16bf;
typedef __attribute__((ext_vector_type(8)))  __bf16 v8bf;
typedef __attribute__((ext_vector_type(8)))  float  v8f;
typedef __attribute__((ext_vector_type(4)))  float  v4f;

__device__ __forceinline__ __bf16 f32_to_bf16(float f) {
    unsigned int u = __builtin_bit_cast(unsigned int, f);
    unsigned int r = u + 0x7fffu + ((u >> 16) & 1u);   // round-to-nearest-even
    unsigned short h = (unsigned short)(r >> 16);
    return __builtin_bit_cast(__bf16, h);
}

__device__ __forceinline__ float sigmoidf_(float v) {
    return 1.0f / (1.0f + expf(-v));
}

// load 8 consecutive f32 and convert to 8 bf16
__device__ __forceinline__ v8bf cvt8(const float* __restrict__ p) {
    v4f f0 = *(const v4f*)p;
    v4f f1 = *(const v4f*)(p + 4);
    v8bf r;
#pragma unroll
    for (int e = 0; e < 4; ++e) { r[e] = f32_to_bf16(f0[e]); r[4 + e] = f32_to_bf16(f1[e]); }
    return r;
}

__device__ __forceinline__ v16bf join16(v8bf a, v8bf b) {
    return __builtin_shufflevector(a, b, 0, 1, 2, 3, 4, 5, 6, 7,
                                   8, 9, 10, 11, 12, 13, 14, 15);
}

// ---------------------------------------------------------------------------
// 3x3 conv (pad=1, stride=1). Activations are bf16 NHWC with a zeroed
// one-pixel spatial border: [B, H+2, W+2, Cpad] (Cpad multiple of 32, channel
// zero-padded). Weights pre-packed bf16 [Cout][tap=9][Cpad].
// GEMM view: C[M=B*H*W, N=Cout], one 16x16 tile per wave.
// Inner loop: 4 unconditional global_load_b128 + 1 WMMA (no masks, no EXEC
// manipulation — the border supplies the conv zero padding).
// Two accumulators (tap parity) shorten the dependent WMMA chain.
// A-frag (ISA 7.12.2): lane%16=row; lane/16 picks K-subset {h*8..+7, 16+h*8..+7}.
// B-frag: lane%16=col; same K-subset as A {h*8..+7, 16+h*8..+7}.
// C/D: VGPR r -> row r + 8*(lane/16), col lane%16.
// ---------------------------------------------------------------------------
__global__ void conv3x3_nhwc_wmma(const __bf16* __restrict__ act, // [B,H+2,W+2,Cpad]
                                  const __bf16* __restrict__ wp,  // [Cout,9,Cpad]
                                  const float*  __restrict__ bias,
                                  float*        __restrict__ y,   // [B,H,W,Cout] f32
                                  int Bn, int H, int W, int Cpad, int Cout) {
    const int lane   = threadIdx.x & 31;
    const int wave   = threadIdx.x >> 5;
    const int tileId = blockIdx.x * (blockDim.x >> 5) + wave;

    const int HW     = H * W;
    const int M      = Bn * HW;
    const int tilesN = Cout >> 4;
    if (tileId >= (M >> 4) * tilesN) return;      // wave-uniform

    const int tm = tileId / tilesN;
    const int tn = tileId - tm * tilesN;
    const int m0 = tm << 4;
    const int n0 = tn << 4;

    const int hgrp = lane >> 4;
    const int l16  = lane & 15;

    const int arow = m0 + l16;                    // output pixel (M index)
    const int ab   = arow / HW;
    const int rem  = arow - ab * HW;
    const int ayy  = rem / W;
    const int axx  = rem - ayy * W;

    const int Hp = H + 2, Wp = W + 2;
    // base at padded position (ayy+1, axx+1)
    const __bf16* rowbase = act + (((size_t)ab * Hp + (ayy + 1)) * Wp + (axx + 1)) * Cpad
                                + hgrp * 8;
    const __bf16* wrow = wp + (size_t)(n0 + l16) * (9 * Cpad) + hgrp * 8;

    v8f acc0 = {}, acc1 = {};
#pragma unroll
    for (int t = 0; t < 9; ++t) {
        const int dy = t / 3 - 1;
        const int dx = t - (t / 3) * 3 - 1;
        const __bf16* pA = rowbase + (ptrdiff_t)(dy * Wp + dx) * Cpad;
        const __bf16* pB = wrow + t * Cpad;
        for (int k0 = 0; k0 < Cpad; k0 += 32) {
            v8bf a0 = *(const v8bf*)(pA + k0);
            v8bf a1 = *(const v8bf*)(pA + k0 + 16);
            v8bf b0 = *(const v8bf*)(pB + k0);
            v8bf b1 = *(const v8bf*)(pB + k0 + 16);
            if (t & 1)
                acc1 = __builtin_amdgcn_wmma_f32_16x16x32_bf16(
                        false, join16(a0, a1), false, join16(b0, b1),
                        (short)0, acc1, false, false);
            else
                acc0 = __builtin_amdgcn_wmma_f32_16x16x32_bf16(
                        false, join16(a0, a1), false, join16(b0, b1),
                        (short)0, acc0, false, false);
        }
    }
#pragma unroll
    for (int r = 0; r < 8; ++r) {
        int row = m0 + r + hgrp * 8;
        y[(size_t)row * Cout + n0 + l16] = acc0[r] + acc1[r] + bias[n0 + l16];
    }
}

// ---------------------------------------------------------------------------
// Fast linear GEMM: C[M,N] = concat(A1[M,K1], A2[M,K2]) * Wp^T + bias.
// A1/A2 f32 (K1, K2 multiples of 32; total K multiple of 64), Wp bf16
// pre-packed [N, K]. M, N multiples of 16. Unrolled x2 with two accumulators.
// act: fused sigmoid.
// ---------------------------------------------------------------------------
__global__ void linear_fast_wmma(const float* __restrict__ A1, int K1,
                                 const float* __restrict__ A2, int K2,
                                 const __bf16* __restrict__ Wp,
                                 const float* __restrict__ bias,
                                 float* __restrict__ C,
                                 int M, int N, int act) {
    const int lane   = threadIdx.x & 31;
    const int wave   = threadIdx.x >> 5;
    const int tileId = blockIdx.x * (blockDim.x >> 5) + wave;

    const int K      = K1 + K2;
    const int tilesN = N >> 4;
    if (tileId >= (M >> 4) * tilesN) return;      // wave-uniform

    const int tm = tileId / tilesN;
    const int tn = tileId - tm * tilesN;
    const int m0 = tm << 4;
    const int n0 = tn << 4;

    const int hgrp = lane >> 4;
    const int l16  = lane & 15;
    const int arow = m0 + l16;

    const float*  r1   = A1 + (size_t)arow * K1;
    const float*  r2   = A2 ? A2 + (size_t)arow * K2 : nullptr;
    const __bf16* wrow = Wp + (size_t)(n0 + l16) * K + hgrp * 8;

    v8f acc0 = {}, acc1 = {};
    for (int k0 = 0; k0 < K; k0 += 64) {          // K % 64 == 0 for all callers
#pragma unroll
        for (int u = 0; u < 2; ++u) {
            int kb  = k0 + u * 32;
            int ka0 = kb + hgrp * 8;
            int ka1 = kb + 16 + hgrp * 8;
            v8bf a0 = (ka0 < K1) ? cvt8(r1 + ka0) : cvt8(r2 + (ka0 - K1));
            v8bf a1 = (ka1 < K1) ? cvt8(r1 + ka1) : cvt8(r2 + (ka1 - K1));
            v8bf b0 = *(const v8bf*)(wrow + kb);
            v8bf b1 = *(const v8bf*)(wrow + kb + 16);
            if (u)
                acc1 = __builtin_amdgcn_wmma_f32_16x16x32_bf16(
                        false, join16(a0, a1), false, join16(b0, b1),
                        (short)0, acc1, false, false);
            else
                acc0 = __builtin_amdgcn_wmma_f32_16x16x32_bf16(
                        false, join16(a0, a1), false, join16(b0, b1),
                        (short)0, acc0, false, false);
        }
    }
#pragma unroll
    for (int r = 0; r < 8; ++r) {
        int row = m0 + r + hgrp * 8;
        float v = acc0[r] + acc1[r] + bias[n0 + l16];
        if (act) v = sigmoidf_(v);
        C[(size_t)row * N + n0 + l16] = v;
    }
}

// ---------------------------------------------------------------------------
// Generic (masked, f32-weight) small WMMA GEMM for the 10-wide readout layers.
// ---------------------------------------------------------------------------
__global__ void linear_small_wmma(const float* __restrict__ A1, int K1,
                                  const float* __restrict__ A2, int K2,
                                  const float* __restrict__ Wt,
                                  const float* __restrict__ bias,
                                  float* __restrict__ C,
                                  int M, int N, int act,
                                  const float* __restrict__ scale_ptr) {
    const int lane   = threadIdx.x & 31;
    const int wave   = threadIdx.x >> 5;
    const int tileId = blockIdx.x * (blockDim.x >> 5) + wave;
    const int K      = K1 + K2;
    const int tilesN = (N + 15) >> 4;
    if (tileId >= ((M + 15) >> 4) * tilesN) return;

    const int tm = tileId / tilesN, tn = tileId - tm * tilesN;
    const int m0 = tm << 4, n0 = tn << 4;
    const int hgrp = lane >> 4, l16 = lane & 15;
    const int arow = m0 + l16;
    const bool arow_ok = arow < M;
    const int bcol = n0 + l16;
    const bool bcol_ok = bcol < N;

    v8f acc = {};
    for (int k0 = 0; k0 < K; k0 += 32) {
        v16bf afrag, bfrag;
#pragma unroll
        for (int v = 0; v < 8; ++v) {
#pragma unroll
            for (int e = 0; e < 2; ++e) {
                int ka = k0 + ((v < 4) ? 0 : 16) + hgrp * 8 + (v & 3) * 2 + e;
                float av = 0.0f;
                if (arow_ok && ka < K)
                    av = (ka < K1) ? A1[(size_t)arow * K1 + ka]
                                   : A2[(size_t)arow * K2 + (ka - K1)];
                afrag[v * 2 + e] = f32_to_bf16(av);
                int kb = k0 + ((v < 4) ? 0 : 16) + hgrp * 8 + (v & 3) * 2 + e;
                float bv = 0.0f;
                if (bcol_ok && kb < K) bv = Wt[(size_t)bcol * K + kb];
                bfrag[v * 2 + e] = f32_to_bf16(bv);
            }
        }
        acc = __builtin_amdgcn_wmma_f32_16x16x32_bf16(
                false, afrag, false, bfrag, (short)0, acc, false, false);
    }
    float sc = 1.0f;
    if (act && scale_ptr) sc = *scale_ptr;
#pragma unroll
    for (int r = 0; r < 8; ++r) {
        int row = m0 + r + hgrp * 8, col = n0 + l16;
        if (row < M && col < N) {
            float v = acc[r] + bias[col];
            if (act) v = sigmoidf_(sc * v);
            C[(size_t)row * N + col] = v;
        }
    }
}

// ---------------------------------------------------------------------------
// Packing kernels (all write the spatially padded bf16 NHWC layout:
// [B, H+2, W+2, Cpad], border = 0)
// ---------------------------------------------------------------------------
// f32 NCHW -> padded bf16 NHWC (channel zero-pad to Cpad, border zero)
__global__ void pack_nchw_nhwc_pad(const float* __restrict__ in, __bf16* __restrict__ out,
                                   int Bn, int C, int H, int W, int Cpad) {
    const int Hp = H + 2, Wp = W + 2;
    int n = Bn * Hp * Wp * Cpad;
    int i = blockIdx.x * blockDim.x + threadIdx.x;
    if (i >= n) return;
    int c = i % Cpad, pix = i / Cpad;
    int xp = pix % Wp, yz = pix / Wp, yp = yz % Hp, b = yz / Hp;
    float v = 0.0f;
    if (c < C && xp >= 1 && xp <= W && yp >= 1 && yp <= H)
        v = in[((size_t)(b * C + c) * H + (yp - 1)) * W + (xp - 1)];
    out[i] = f32_to_bf16(v);
}

// f32 [Cout,Cin,3,3] -> bf16 [Cout][tap][Cpad], zero-pad channels
__global__ void pack_conv_w(const float* __restrict__ w, __bf16* __restrict__ out,
                            int Cout, int Cin, int Cpad) {
    int n = Cout * 9 * Cpad;
    int i = blockIdx.x * blockDim.x + threadIdx.x;
    if (i >= n) return;
    int ci = i % Cpad, t = (i / Cpad) % 9, co = i / (9 * Cpad);
    float v = (ci < Cin) ? w[((size_t)co * Cin + ci) * 9 + t] : 0.0f;
    out[i] = f32_to_bf16(v);
}

__global__ void pack_f32_bf16(const float* __restrict__ in, __bf16* __restrict__ out, int n) {
    int i = blockIdx.x * blockDim.x + threadIdx.x;
    if (i < n) out[i] = f32_to_bf16(in[i]);
}

// ---------------------------------------------------------------------------
// BatchNorm (training stats, biased var) over NHWC: coalesced grid-stride
// reads, per-block LDS f32 atomics, global f32 atomic partials, finalize.
// ---------------------------------------------------------------------------
__global__ void bn_zero(float* __restrict__ sums, int n) {
    int i = blockIdx.x * blockDim.x + threadIdx.x;
    if (i < n) sums[i] = 0.0f;
}

__global__ void bn_partial(const float* __restrict__ x, float* __restrict__ sums,
                           int n, int C) {
    __shared__ float ls[256], lq[256];
    int tid = threadIdx.x;
    for (int j = tid; j < C; j += 256) { ls[j] = 0.0f; lq[j] = 0.0f; }
    __syncthreads();
    int stride = gridDim.x * blockDim.x;
    for (int i = blockIdx.x * blockDim.x + tid; i < n; i += stride) {
        float v = x[i];
        int c = i % C;
        atomicAdd(&ls[c], v);
        atomicAdd(&lq[c], v * v);
    }
    __syncthreads();
    for (int j = tid; j < C; j += 256) {
        atomicAdd(&sums[j], ls[j]);
        atomicAdd(&sums[C + j], lq[j]);
    }
}

__global__ void bn_finalize(const float* __restrict__ sums, float* __restrict__ stats,
                            int C, float inv_total) {
    int c = blockIdx.x * blockDim.x + threadIdx.x;
    if (c >= C) return;
    float mean = sums[c] * inv_total;
    float var  = sums[C + c] * inv_total - mean * mean;
    stats[2 * c + 0] = mean;
    stats[2 * c + 1] = rsqrtf(var + 1e-5f);
}

__global__ void bn_apply(const float* __restrict__ in, float* __restrict__ out,
                         const float* __restrict__ stats,
                         const float* __restrict__ g, const float* __restrict__ beta,
                         int C, int n, int act) {
    int i = blockIdx.x * blockDim.x + threadIdx.x;
    if (i >= n) return;
    int c = i % C;
    float v = (in[i] - stats[2 * c]) * stats[2 * c + 1] * g[c] + beta[c];
    if (act) v = sigmoidf_(v);
    out[i] = v;
}

// 2x2/2 max pool, NHWC (unpadded f32 in/out)
__global__ void maxpool_nhwc(const float* __restrict__ in, float* __restrict__ out,
                             int Bn, int H, int W, int C) {
    const int Ho = H >> 1, Wo = W >> 1;
    const int n = Bn * Ho * Wo * C;
    int i = blockIdx.x * blockDim.x + threadIdx.x;
    if (i >= n) return;
    int c = i % C, pix = i / C;
    int x = pix % Wo, yz = pix / Wo, y = yz % Ho, b = yz / Ho;
    const float* p = in + ((size_t)(b * H + 2 * y) * W + 2 * x) * C + c;
    size_t rs = (size_t)W * C;
    out[i] = fmaxf(fmaxf(p[0], p[C]), fmaxf(p[rs], p[rs + C]));
}

// padded bf16 NHWC = cx(NHWC f32, unpadded) + state(NCHW f32); border zero
__global__ void add_pack_pad(const float* __restrict__ cxN, const float* __restrict__ stN,
                             __bf16* __restrict__ out, int Bn, int C, int H, int W) {
    const int Hp = H + 2, Wp = W + 2;
    int n = Bn * Hp * Wp * C;
    int i = blockIdx.x * blockDim.x + threadIdx.x;
    if (i >= n) return;
    int c = i % C, pix = i / C;
    int xp = pix % Wp, yz = pix / Wp, yp = yz % Hp, b = yz / Hp;
    float v = 0.0f;
    if (xp >= 1 && xp <= W && yp >= 1 && yp <= H) {
        int x = xp - 1, y = yp - 1;
        size_t nhwc = ((size_t)(b * H + y) * W + x) * C + c;
        size_t nchw = ((size_t)(b * C + c) * H + y) * W + x;
        v = cxN[nhwc] + stN[nchw];
    }
    out[i] = f32_to_bf16(v);
}

// adaptive-threshold spike update for conv cells (only spike is live).
// cx/tauA/tauM are unpadded NHWC f32; mem/spk/b are NCHW f32 (harness layout).
// Iterates the PADDED volume; border writes zero spike (bf16 out only).
// Outputs: padded bf16 NHWC (feeds next conv) and/or f32 NCHW (feeds l1).
__global__ void mem_update_conv(const float* __restrict__ cxN,
                                const float* __restrict__ mem, const float* __restrict__ spk,
                                const float* __restrict__ bb,
                                const float* __restrict__ tauA, const float* __restrict__ tauM,
                                __bf16* __restrict__ spk_bf, float* __restrict__ spk_f32,
                                int Bn, int C, int H, int W) {
    const int Hp = H + 2, Wp = W + 2;
    int n = Bn * Hp * Wp * C;
    int i = blockIdx.x * blockDim.x + threadIdx.x;
    if (i >= n) return;
    int c = i % C, pix = i / C;
    int xp = pix % Wp, yz = pix / Wp, yp = yz % Hp, b = yz / Hp;
    bool interior = (xp >= 1) & (xp <= W) & (yp >= 1) & (yp <= H);
    float s = 0.0f;
    if (interior) {
        int x = xp - 1, y = yp - 1;
        size_t nhwc = ((size_t)(b * H + y) * W + x) * C + c;
        size_t nchw = ((size_t)(b * C + c) * H + y) * W + x;
        float ta = tauA[nhwc];
        float b2 = ta * bb[nchw] + (1.0f - ta) * spk[nchw];
        float m  = mem[nchw];
        float m2 = m + (cxN[nhwc] - m) * tauM[nhwc];
        s = ((m2 - (0.1f + 1.8f * b2)) > 0.0f) ? 1.0f : 0.0f;
        if (spk_f32) spk_f32[nchw] = s;
    }
    if (spk_bf) spk_bf[i] = f32_to_bf16(s);
}

// flat f32 spike update for the recurrent linear cells
__global__ void mem_update_flat(const float* __restrict__ cx, const float* __restrict__ mem,
                                const float* __restrict__ spk, const float* __restrict__ bb,
                                const float* __restrict__ tauA, const float* __restrict__ tauM,
                                float* __restrict__ spk_out, int n) {
    int i = blockIdx.x * blockDim.x + threadIdx.x;
    if (i >= n) return;
    float ta = tauA[i];
    float b2 = ta * bb[i] + (1.0f - ta) * spk[i];
    float m2 = mem[i] + (cx[i] - mem[i]) * tauM[i];
    spk_out[i] = ((m2 - (0.1f + 1.8f * b2)) > 0.0f) ? 1.0f : 0.0f;
}

// leaky-integrator readout + log_softmax (N=10)
__global__ void readout_kernel(const float* __restrict__ d3, const float* __restrict__ memo,
                               const float* __restrict__ tauM2, float* __restrict__ out,
                               int Bn, int N) {
    int r = blockIdx.x * blockDim.x + threadIdx.x;
    if (r >= Bn) return;
    float mo[16];
    float mx = -1e30f;
    for (int j = 0; j < N; ++j) {
        float m = memo[r * N + j];
        float v = m + (d3[r * N + j] - m) * tauM2[r * N + j];
        mo[j] = v;
        mx = fmaxf(mx, v);
    }
    float s = 0.0f;
    for (int j = 0; j < N; ++j) s += expf(mo[j] - mx);
    float lse = mx + logf(s);
    for (int j = 0; j < N; ++j) out[r * N + j] = mo[j] - lse;
}

// ---------------------------------------------------------------------------
// Host orchestration
// ---------------------------------------------------------------------------
extern "C" void kernel_launch(void* const* d_in, const int* in_sizes, int n_in,
                              void* d_out, int out_size, void* d_ws, size_t ws_size,
                              hipStream_t stream) {
    const int Bn = 256;

    const float* inputs = (const float*)d_in[0];
    const float* st[16];
    for (int i = 0; i < 16; ++i) st[i] = (const float*)d_in[1 + i];
    // st: 0..2 mem/spk/b c1 | 3..5 c2 | 6..8 c3 | 9..11 l1 | 12..14 l2 | 15 mem_o
    const float* const* P = (const float* const*)(d_in + 17);
    // P: [0..11] c1 (wx,bx,wm,bm,wa,ba,g1,b1,g2,b2,g3,b3); [12..23] c2; [24..35] c3
    //    [36..41] l1 (wx,bx,wm,bm,wa,ba); [42..47] l2
    //    [48] w3, [49] b3, [50] wm3, [51] bm3, [52] alpha
    (void)in_sizes; (void)n_in; (void)out_size; (void)ws_size;

    float* ws = (float*)d_ws;
    size_t off = 0;
    auto alloc = [&](size_t nf) { float* p = ws + off; off += (nf + 255) & ~(size_t)255; return p; };
    float*  raw   = alloc(25690112);                 // f32 NHWC conv out (max c2@28x28x128)
    float*  cx    = alloc(6422528);                  // f32 NHWC (unpadded)
    float*  tmb   = alloc(6422528);                  // tauM (f32 NHWC)
    float*  tab   = alloc(6422528);                  // tauA (f32 NHWC)
    __bf16* actIn = (__bf16*)alloc(3686400);         // padded bf16 [256,30,30,32]
    __bf16* tpack = (__bf16*)alloc(4194304);         // padded gate input (max [256,16,16,128])
    __bf16* spkA  = (__bf16*)alloc(3686400);         // spike padded bf16 (c1 out, 30x30x32)
    __bf16* spkB  = (__bf16*)alloc(4194304);         // spike padded bf16 (c2 out, 16x16x128)
    float*  spk3  = alloc(3211264);                  // spike f32 NCHW (c3 out, feeds l1)
    __bf16* wpk   = (__bf16*)alloc(3211264);         // packed weights (max l1_wx 6.4M bf16)
    float*  sums  = alloc(1024);
    float*  stats = alloc(1024);
    float*  dx    = alloc(256 * 512);
    float*  zm    = alloc(256 * 512);
    float*  za    = alloc(256 * 512);
    float*  s1    = alloc(256 * 512);
    float*  s2    = alloc(256 * 512);
    float*  d3    = alloc(256 * 16);
    float*  tM2   = alloc(256 * 16);

    auto run_conv = [&](const __bf16* act, const float* w, const float* bias, float* y,
                        int Cin, int Cpad, int H, int W, int Cout) {
        int nw = Cout * 9 * Cpad;
        pack_conv_w<<<(nw + 255) / 256, 256, 0, stream>>>(w, wpk, Cout, Cin, Cpad);
        int tiles = ((Bn * H * W) / 16) * (Cout / 16);
        conv3x3_nhwc_wmma<<<(tiles + 7) / 8, 256, 0, stream>>>(act, wpk, bias, y,
                                                               Bn, H, W, Cpad, Cout);
    };
    auto run_bn = [&](const float* in, float* out, const float* g, const float* be,
                      int C, int HW, int act) {
        int n = Bn * C * HW;
        bn_zero<<<(2 * C + 255) / 256, 256, 0, stream>>>(sums, 2 * C);
        int nb = (n + 255) / 256; if (nb > 2048) nb = 2048;
        bn_partial<<<nb, 256, 0, stream>>>(in, sums, n, C);
        bn_finalize<<<(C + 255) / 256, 256, 0, stream>>>(sums, stats, C, 1.0f / (float)(Bn * HW));
        bn_apply<<<(n + 255) / 256, 256, 0, stream>>>(in, out, stats, g, be, C, n, act);
    };
    auto run_lin = [&](const float* A1, int K1, const float* A2, int K2,
                       const float* Wf, const float* bias, float* Cm, int N, int act) {
        int K = K1 + K2, nw = N * K;
        pack_f32_bf16<<<(nw + 255) / 256, 256, 0, stream>>>(Wf, wpk, nw);
        int tiles = (Bn / 16) * (N / 16);
        linear_fast_wmma<<<(tiles + 7) / 8, 256, 0, stream>>>(A1, K1, A2, K2, wpk, bias,
                                                              Cm, Bn, N, act);
    };

    auto conv_cell = [&](const __bf16* actx, int Cin, int Cpad,
                         const float* mem, const float* spk, const float* bb,
                         const float* const* pp, int Cout, int H, int W, bool pool,
                         __bf16* spk_bf, float* spk_f32) {
        run_conv(actx, pp[0], pp[1], raw, Cin, Cpad, H, W, Cout);
        int h = pool ? H / 2 : H, wd = pool ? W / 2 : W;
        if (pool) {
            run_bn(raw, raw, pp[6], pp[7], Cout, H * W, 0);
            int np = Bn * h * wd * Cout;
            maxpool_nhwc<<<(np + 255) / 256, 256, 0, stream>>>(raw, cx, Bn, H, W, Cout);
        } else {
            run_bn(raw, cx, pp[6], pp[7], Cout, H * W, 0);
        }
        int npad = Bn * (h + 2) * (wd + 2) * Cout, pb = (npad + 255) / 256;
        add_pack_pad<<<pb, 256, 0, stream>>>(cx, mem, tpack, Bn, Cout, h, wd);
        run_conv(tpack, pp[2], pp[3], raw, Cout, Cout, h, wd, Cout);
        run_bn(raw, tmb, pp[8], pp[9], Cout, h * wd, 1);         // tauM
        add_pack_pad<<<pb, 256, 0, stream>>>(cx, bb, tpack, Bn, Cout, h, wd);
        run_conv(tpack, pp[4], pp[5], raw, Cout, Cout, h, wd, Cout);
        run_bn(raw, tab, pp[10], pp[11], Cout, h * wd, 1);       // tauA
        mem_update_conv<<<pb, 256, 0, stream>>>(cx, mem, spk, bb, tab, tmb,
                                                spk_bf, spk_f32, Bn, Cout, h, wd);
    };

    // --- pack network input: [B,1,28,28] f32 -> padded [B,30,30,32] bf16 ---
    {
        int n = Bn * 30 * 30 * 32;
        pack_nchw_nhwc_pad<<<(n + 255) / 256, 256, 0, stream>>>(inputs, actIn, Bn, 1, 28, 28, 32);
    }

    // --- conv stack ---
    conv_cell(actIn, 1, 32,  st[0], st[1], st[2], P + 0,  32, 28, 28, false, spkA, nullptr);
    conv_cell(spkA, 32, 32,  st[3], st[4], st[5], P + 12, 128, 28, 28, true,  spkB, nullptr);
    conv_cell(spkB, 128, 128, st[6], st[7], st[8], P + 24, 256, 14, 14, true, nullptr, spk3);

    // --- recurrent linear cells (F = 12544, H1 = 512) ---
    const int H1 = 512, F = 12544;
    int nl = Bn * H1, elb = (nl + 255) / 256;
    run_lin(spk3, F, nullptr, 0, P[36], P[37], dx, H1, 0);
    run_lin(dx, H1, st[9],  H1, P[38], P[39], zm, H1, 1);        // tauM
    run_lin(dx, H1, st[11], H1, P[40], P[41], za, H1, 1);        // tauA
    mem_update_flat<<<elb, 256, 0, stream>>>(dx, st[9], st[10], st[11], za, zm, s1, nl);
    run_lin(s1, H1, nullptr, 0, P[42], P[43], dx, H1, 0);
    run_lin(dx, H1, st[12], H1, P[44], P[45], zm, H1, 1);
    run_lin(dx, H1, st[14], H1, P[46], P[47], za, H1, 1);
    mem_update_flat<<<elb, 256, 0, stream>>>(dx, st[12], st[13], st[14], za, zm, s2, nl);

    // --- readout (N=10): generic masked WMMA path ---
    const int O = 10;
    {
        int tiles = ((Bn + 15) / 16) * ((O + 15) / 16);
        linear_small_wmma<<<(tiles + 7) / 8, 256, 0, stream>>>(
            s2, H1, nullptr, 0, P[48], P[49], d3, Bn, O, 0, nullptr);
        linear_small_wmma<<<(tiles + 7) / 8, 256, 0, stream>>>(
            d3, O, st[15], O, P[50], P[51], tM2, Bn, O, 1, P[52]);
    }
    readout_kernel<<<1, 256, 0, stream>>>(d3, st[15], tM2, (float*)d_out, Bn, O);
}
